// Bottleneck_DCNV4_18640158064726
// MI455X (gfx1250) — matrix operand order repaired and checked by reading, not compiled, not run on hardware
//
#include <hip/hip_runtime.h>
#include <hip/hip_bf16.h>

// ---------------------------------------------------------------------------
// Bottleneck_DCNV4 for MI455X (gfx1250, wave32, WMMA).
//
// Roofline: ~24 GFLOP, dominated by cv1 (15.1 GF, K=2304 implicit GEMM).
// All four GEMM stages use v_wmma_f32_16x16x32_f16 (f16 A/B, f32 accum),
// register-blocked 4 N-tiles per wave so each A fragment feeds 4 independent
// wmma (4x less A traffic, better XDL pipelining).
// cv1 uses tap-major K ordering (k = tap*256 + c): bounds checks / spatial
// offsets are per-tap loop invariants, all fragment loads contiguous b128
// (x pre-transposed to NHWC, w1 pre-packed to f16 tap-major).
// DCNv3 gather core runs on NHWC so the 256-channel axis coalesces.
// ---------------------------------------------------------------------------

typedef __attribute__((ext_vector_type(16))) _Float16 v16h;
typedef __attribute__((ext_vector_type(8)))  float    v8f;

#define NB   4
#define C1c  256
#define CHc  128
#define C2c  256
#define Hc   80
#define Wc   80
#define HWc  6400
#define MPIX (NB*HWc)      // 25600 pixels
#define K1c  (C1c*9)       // 2304
#define M_T  (MPIX/16)     // 1600 M tiles

// WMMA fragment layouts (CDNA5 ISA §7.12.2, wave32):
//   A (16x32 f16): lane L: M = L&15 ; half h = L>>4 ; element i ->
//                  K = (i<8 ? 0 : 16) + 8*h + (i&7)   (two contiguous runs of 8)
//   B (32x16 f16): lane L: N = L&15 ; half h = L>>4 ; element i -> K = 16*h + i in this code (A loaded alike)
//   C/D (16x16 f32, 8 VGPRs): lane L: N = L&15 ; vgpr r -> M = r + 8*(L>>4)
__device__ __forceinline__ float silu_f(float v) {
  return v / (1.0f + expf(-v));
}

// ---------------------------------------------------------------------------
// Prep kernel A: x (NCHW) -> xnhwc (N, HW, C) via LDS-tiled 32x32 transpose.
// ---------------------------------------------------------------------------
__global__ void __launch_bounds__(256)
k_x_to_nhwc(const float* __restrict__ x, float* __restrict__ xnhwc)
{
  __shared__ float tile[32][33];
  const int hw0 = blockIdx.x * 32;
  const int c0  = blockIdx.y * 32;
  const int ni  = blockIdx.z;
  const int tx  = threadIdx.x & 31;
  const int ty  = threadIdx.x >> 5;          // 0..7
  const float* src = x + (size_t)ni * C1c * HWc;
#pragma unroll
  for (int k = 0; k < 4; ++k) {
    int c = c0 + ty + k * 8;
    tile[ty + k * 8][tx] = src[(size_t)c * HWc + hw0 + tx];
  }
  __syncthreads();
  float* dst = xnhwc + (size_t)ni * HWc * C1c;
#pragma unroll
  for (int k = 0; k < 4; ++k) {
    int hw = hw0 + ty + k * 8;
    dst[(size_t)hw * C1c + c0 + tx] = tile[tx][ty + k * 8];
  }
}

// ---------------------------------------------------------------------------
// Prep kernel B: w1 (O,I,3,3 f32, k = c*9+t) -> wpk (O, t, c f16, k = t*256+c)
// ---------------------------------------------------------------------------
__global__ void __launch_bounds__(256)
k_pack_w1(const float* __restrict__ w1, _Float16* __restrict__ wpk)
{
  int idx = blockIdx.x * 256 + threadIdx.x;          // 128*2304 = 294912
  if (idx >= CHc * K1c) return;
  int oc = idx / K1c;
  int r  = idx - oc * K1c;
  int c  = r / 9;
  int t  = r - c * 9;
  wpk[(size_t)oc * K1c + t * C1c + c] = (_Float16)w1[idx];
}

// ---------------------------------------------------------------------------
// Kernel 1: cv1 = 3x3 conv C1->CH implicit GEMM (M=25600, N=128, K=2304)
//           + BN + SiLU.  4 N-tiles per wave (N group = 64 oc).
//           Output: f16 row-major [m][oc] (feeds cv2's A directly).
// ---------------------------------------------------------------------------
__global__ void __launch_bounds__(256)
k_cv1(const float* __restrict__ xnhwc, const _Float16* __restrict__ wpk,
      const float* __restrict__ g1, const float* __restrict__ b1,
      const float* __restrict__ m1, const float* __restrict__ v1,
      _Float16* __restrict__ y1mk)
{
  const int lane = threadIdx.x & 31;
  const int wave = blockIdx.x * (blockDim.x >> 5) + (threadIdx.x >> 5);
  const int tm = wave % M_T;
  const int tg = wave / M_T;                 // 0..1  (4 tiles of 16 oc each)
  if (tg >= CHc / 64) return;                // wave-uniform
  const int hi = lane >> 4, ln = lane & 15;

  // M tile lies inside one image (6400 % 16 == 0)
  const int ni  = (tm * 16) / HWc;
  const int hwb = tm * 16 - ni * HWc;
  const int hw  = hwb + ln;
  const int h0  = hw / Wc, w0 = hw - (hw / Wc) * Wc;
  const float* xb = xnhwc + (size_t)ni * HWc * C1c;

  const _Float16* brow[4];
#pragma unroll
  for (int j = 0; j < 4; ++j)
    brow[j] = wpk + (size_t)((tg * 4 + j) * 16 + ln) * K1c;

  // per-tap spatial offset + validity (K-loop invariants)
  int   offs[9];
  float vmsk[9];
#pragma unroll
  for (int t = 0; t < 9; ++t) {
    int hh = h0 + t / 3 - 1, ww = w0 + t % 3 - 1;
    bool v = ((unsigned)hh < (unsigned)Hc) && ((unsigned)ww < (unsigned)Wc);
    offs[t] = v ? (hh * Wc + ww) : 0;        // clamped safe address
    vmsk[t] = v ? 1.0f : 0.0f;
  }

  const v8f zero = {};
  v8f acc[4] = {zero, zero, zero, zero};
  for (int t = 0; t < 9; ++t) {
    const float* ab = xb + (size_t)offs[t] * C1c;
    const float  vm = vmsk[t];
    const int    toff = t * C1c;
#pragma unroll 2
    for (int cb = 0; cb < C1c; cb += 32) {
      v16h a;
      const float* ap = ab + cb + (hi << 4);
      __builtin_prefetch(ap + 64, 0, 1);     // global_prefetch_b8
#pragma unroll
      for (int i = 0; i < 8; ++i) {
        a[i]     = (_Float16)(vm * ap[i]);
        a[i + 8] = (_Float16)(vm * ap[i + 8]);
      }
#pragma unroll
      for (int j = 0; j < 4; ++j) {
        v16h b;
        const _Float16* bp = brow[j] + toff + cb + (hi << 4);
#pragma unroll
        for (int i = 0; i < 16; ++i) b[i] = bp[i];
        acc[j] = __builtin_amdgcn_wmma_f32_16x16x32_f16(false, a, false, b,
                                                        (short)0, acc[j], false, false);
      }
    }
  }
#pragma unroll
  for (int j = 0; j < 4; ++j) {
    const int oc = (tg * 4 + j) * 16 + ln;
    const float sc = g1[oc] * rsqrtf(v1[oc] + 1e-5f);
    const float sh = b1[oc] - m1[oc] * sc;
#pragma unroll
    for (int r = 0; r < 8; ++r) {
      int mo = tm * 16 + r + (hi << 3);
      float v = acc[j][r] * sc + sh;
      y1mk[(size_t)mo * CHc + oc] = (_Float16)silu_f(v);
    }
  }
}

// ---------------------------------------------------------------------------
// Kernel 2: cv2 = 1x1 conv CH->C2 GEMM (M=25600, N=256, K=128) + BN + SiLU.
//           4 N-tiles per wave.  Output: f32 NHWC.
// ---------------------------------------------------------------------------
__global__ void __launch_bounds__(256)
k_cv2(const _Float16* __restrict__ y1mk, const float* __restrict__ w2,
      const float* __restrict__ g2, const float* __restrict__ b2,
      const float* __restrict__ m2, const float* __restrict__ v2,
      float* __restrict__ ynhwc)
{
  const int lane = threadIdx.x & 31;
  const int wave = blockIdx.x * (blockDim.x >> 5) + (threadIdx.x >> 5);
  const int tm = wave % M_T;
  const int tg = wave / M_T;                 // 0..3
  if (tg >= C2c / 64) return;
  const int hi = lane >> 4, ln = lane & 15;

  const _Float16* arow = y1mk + (size_t)(tm * 16 + ln) * CHc;
  const float* brow[4];
#pragma unroll
  for (int j = 0; j < 4; ++j)
    brow[j] = w2 + (size_t)((tg * 4 + j) * 16 + ln) * CHc;

  const v8f zero = {};
  v8f acc[4] = {zero, zero, zero, zero};
#pragma unroll 2
  for (int k0 = 0; k0 < CHc; k0 += 32) {
    v16h a;
    const _Float16* ap = arow + k0 + (hi << 4);
#pragma unroll
    for (int i = 0; i < 8; ++i) { a[i] = ap[i]; a[i + 8] = ap[i + 8]; }
#pragma unroll
    for (int j = 0; j < 4; ++j) {
      v16h b;
      const float* bp = brow[j] + k0 + (hi << 4);
#pragma unroll
      for (int i = 0; i < 16; ++i) b[i] = (_Float16)bp[i];
      acc[j] = __builtin_amdgcn_wmma_f32_16x16x32_f16(false, a, false, b,
                                                      (short)0, acc[j], false, false);
    }
  }
#pragma unroll
  for (int j = 0; j < 4; ++j) {
    const int oc = (tg * 4 + j) * 16 + ln;
    const float sc = g2[oc] * rsqrtf(v2[oc] + 1e-5f);
    const float sh = b2[oc] - m2[oc] * sc;
#pragma unroll
    for (int r = 0; r < 8; ++r) {
      int mo = tm * 16 + r + (hi << 3);
      float v = acc[j][r] * sc + sh;
      ynhwc[(size_t)mo * C2c + oc] = silu_f(v);
    }
  }
}

// ---------------------------------------------------------------------------
// Kernel 3: depthwise 3x3 + bias + LayerNorm + exact GELU, fused with the
//           offset(18)/mask(9 softmax) projections.  One block per pixel;
//           the 27 GEMV dots are split 8 ways + LDS partial reduction.
// ---------------------------------------------------------------------------
__global__ void __launch_bounds__(256)
k_dw_ln_gelu(const float* __restrict__ ynhwc,
             const float* __restrict__ dw_w, const float* __restrict__ dw_b,
             const float* __restrict__ ln_g, const float* __restrict__ ln_b,
             const float* __restrict__ off_w, const float* __restrict__ off_b,
             const float* __restrict__ msk_w, const float* __restrict__ msk_b,
             float* __restrict__ off_out, float* __restrict__ msk_out)
{
  __shared__ float sred[256];
  __shared__ float sdw[256];
  __shared__ float spart[27][8];
  __shared__ float slog[9];
  const int pix = blockIdx.x;
  const int c   = threadIdx.x;
  const int ni  = pix / HWc;
  const int hw  = pix - ni * HWc;
  const int h   = hw / Wc, w = hw - (hw / Wc) * Wc;
  const float* base = ynhwc + (size_t)ni * HWc * C2c;

  float acc = dw_b[c];
#pragma unroll
  for (int t = 0; t < 9; ++t) {
    int dy = t / 3 - 1, dx = t % 3 - 1;
    int hh = h + dy, ww = w + dx;
    if ((unsigned)hh < (unsigned)Hc && (unsigned)ww < (unsigned)Wc)
      acc += base[(size_t)(hh * Wc + ww) * C2c + c] * dw_w[c * 9 + t];
  }
  // LayerNorm over channels
  sred[c] = acc;  __syncthreads();
  for (int s = 128; s > 0; s >>= 1) { if (c < s) sred[c] += sred[c + s]; __syncthreads(); }
  const float mu = sred[0] * (1.0f / 256.0f);
  __syncthreads();
  const float d = acc - mu;
  sred[c] = d * d;  __syncthreads();
  for (int s = 128; s > 0; s >>= 1) { if (c < s) sred[c] += sred[c + s]; __syncthreads(); }
  const float var = sred[0] * (1.0f / 256.0f);
  float vn = d * rsqrtf(var + 1e-6f) * ln_g[c] + ln_b[c];
  vn = 0.5f * vn * (1.0f + erff(vn * 0.70710678118654752f));   // exact GELU
  sdw[c] = vn;
  __syncthreads();

  // 27 dots (18 offset + 9 mask logits), each split across 8 threads
  const int j = c >> 3, l = c & 7;
  if (j < 27) {
    const float* wr = (j < 18) ? (off_w + j * C2c) : (msk_w + (j - 18) * C2c);
    const int k0 = l * 32;
    float p = 0.0f;
#pragma unroll
    for (int k = 0; k < 32; ++k) p += sdw[k0 + k] * wr[k0 + k];
    spart[j][l] = p;
  }
  __syncthreads();
  if (c < 27) {
    float o = (c < 18) ? off_b[c] : msk_b[c - 18];
#pragma unroll
    for (int l2 = 0; l2 < 8; ++l2) o += spart[c][l2];
    if (c < 18) off_out[(size_t)pix * 18 + c] = o;
    else        slog[c - 18] = o;
  }
  __syncthreads();
  if (c < 9) {                               // softmax over K=9
    float mx = slog[0];
#pragma unroll
    for (int jj = 1; jj < 9; ++jj) mx = fmaxf(mx, slog[jj]);
    float sum = 0.0f;
#pragma unroll
    for (int jj = 0; jj < 9; ++jj) sum += expf(slog[jj] - mx);
    msk_out[(size_t)pix * 9 + c] = expf(slog[c] - mx) / sum;
  }
}

// ---------------------------------------------------------------------------
// Shared 256x256 NHWC GEMM core (input_proj / output_proj), 4 N-tiles/wave.
//   out[m][j] = sum_k A[m*256+k] * Wt[j*256+k]
// ---------------------------------------------------------------------------
__device__ __forceinline__ void gemm256_nhwc4(const float* __restrict__ A,
                                              const float* __restrict__ Wt,
                                              int tm, int tg, int lane, v8f acc[4])
{
  const int hi = lane >> 4, ln = lane & 15;
  const float* arow = A + (size_t)(tm * 16 + ln) * C2c;
  const float* brow[4];
#pragma unroll
  for (int j = 0; j < 4; ++j)
    brow[j] = Wt + (size_t)((tg * 4 + j) * 16 + ln) * C2c;
#pragma unroll 2
  for (int k0 = 0; k0 < C2c; k0 += 32) {
    v16h a;
    const float* ap = arow + k0 + (hi << 4);
    __builtin_prefetch(ap + 64, 0, 1);
#pragma unroll
    for (int i = 0; i < 8; ++i) { a[i] = (_Float16)ap[i]; a[i + 8] = (_Float16)ap[i + 8]; }
#pragma unroll
    for (int j = 0; j < 4; ++j) {
      v16h b;
      const float* bp = brow[j] + k0 + (hi << 4);
#pragma unroll
      for (int i = 0; i < 16; ++i) b[i] = (_Float16)bp[i];
      acc[j] = __builtin_amdgcn_wmma_f32_16x16x32_f16(false, a, false, b,
                                                      (short)0, acc[j], false, false);
    }
  }
}

// Kernel 4: input_proj (M=25600, N=256, K=256) + bias -> f32 NHWC
__global__ void __launch_bounds__(256)
k_inproj(const float* __restrict__ ynhwc, const float* __restrict__ in_w,
         const float* __restrict__ in_b, float* __restrict__ xproj)
{
  const int lane = threadIdx.x & 31;
  const int wave = blockIdx.x * (blockDim.x >> 5) + (threadIdx.x >> 5);
  const int tm = wave % M_T, tg = wave / M_T;
  if (tg >= C2c / 64) return;
  const v8f zero = {};
  v8f acc[4] = {zero, zero, zero, zero};
  gemm256_nhwc4(ynhwc, in_w, tm, tg, lane, acc);
  const int hi = lane >> 4, ln = lane & 15;
#pragma unroll
  for (int j = 0; j < 4; ++j) {
    const int oc = (tg * 4 + j) * 16 + ln;
    const float bias = in_b[oc];
#pragma unroll
    for (int r = 0; r < 8; ++r) {
      int mo = tm * 16 + r + (hi << 3);
      xproj[(size_t)mo * C2c + oc] = acc[j][r] + bias;
    }
  }
}

// ---------------------------------------------------------------------------
// Kernel 5: DCNv3 core.  Padded coords: px = w + p/3 + off_x,
//           py = h + p%3 + off_y.  Sample nonzero iff 1<=xi<=80 && 1<=yi<=80.
// ---------------------------------------------------------------------------
__device__ __forceinline__ float fetch_pad(const float* __restrict__ xb,
                                           int yy, int xx, int c)
{
  if ((unsigned)(xx - 1) < (unsigned)Wc && (unsigned)(yy - 1) < (unsigned)Hc)
    return xb[(size_t)((yy - 1) * Wc + (xx - 1)) * C2c + c];
  return 0.0f;
}

__global__ void __launch_bounds__(256)
k_dcnv3(const float* __restrict__ xproj, const float* __restrict__ off_in,
        const float* __restrict__ msk_in, float* __restrict__ z)
{
  __shared__ float soff[18];
  __shared__ float smask[9];
  const int pix = blockIdx.x;
  const int c   = threadIdx.x;
  if (c < 18) soff[c]  = off_in[(size_t)pix * 18 + c];
  if (c < 9)  smask[c] = msk_in[(size_t)pix * 9 + c];
  __syncthreads();
  const int ni = pix / HWc;
  const int hw = pix - ni * HWc;
  const int h  = hw / Wc, w = hw - (hw / Wc) * Wc;
  const float* xb = xproj + (size_t)ni * HWc * C2c;

  float acc = 0.0f;
#pragma unroll
  for (int p = 0; p < 9; ++p) {
    float px = (float)(w + p / 3) + soff[p * 2 + 0];
    float py = (float)(h + p % 3) + soff[p * 2 + 1];
    float fx = floorf(px), fy = floorf(py);
    float lw = px - fx,    lh = py - fy;
    int x0 = (int)fx, y0 = (int)fy;
    float v00 = fetch_pad(xb, y0,     x0,     c);
    float v01 = fetch_pad(xb, y0,     x0 + 1, c);
    float v10 = fetch_pad(xb, y0 + 1, x0,     c);
    float v11 = fetch_pad(xb, y0 + 1, x0 + 1, c);
    acc += smask[p] * ((1.0f - lh) * (1.0f - lw) * v00 + (1.0f - lh) * lw * v01
                     + lh * (1.0f - lw) * v10 + lh * lw * v11);
  }
  z[(size_t)pix * C2c + c] = acc;
}

// ---------------------------------------------------------------------------
// Kernel 6: output_proj GEMM + bias + BN + SiLU + residual -> f32 NCHW out
// ---------------------------------------------------------------------------
__global__ void __launch_bounds__(256)
k_outproj(const float* __restrict__ z, const float* __restrict__ out_w,
          const float* __restrict__ out_b,
          const float* __restrict__ g3, const float* __restrict__ b3,
          const float* __restrict__ m3, const float* __restrict__ v3,
          const float* __restrict__ xin, float* __restrict__ out)
{
  const int lane = threadIdx.x & 31;
  const int wave = blockIdx.x * (blockDim.x >> 5) + (threadIdx.x >> 5);
  const int tm = wave % M_T, tg = wave / M_T;
  if (tg >= C2c / 64) return;
  const v8f zero = {};
  v8f acc[4] = {zero, zero, zero, zero};
  gemm256_nhwc4(z, out_w, tm, tg, lane, acc);
  const int hi = lane >> 4, ln = lane & 15;
  const int ni  = (tm * 16) / HWc;           // tile within one image
  const int hwb = tm * 16 - ni * HWc;
#pragma unroll
  for (int j = 0; j < 4; ++j) {
    const int oc = (tg * 4 + j) * 16 + ln;
    const float sc   = g3[oc] * rsqrtf(v3[oc] + 1e-5f);
    const float sh   = b3[oc] - m3[oc] * sc;
    const float bias = out_b[oc];
#pragma unroll
    for (int r = 0; r < 8; ++r) {
      int hwo = hwb + r + (hi << 3);
      float v = (acc[j][r] + bias) * sc + sh;
      v = silu_f(v);
      size_t idx = (size_t)(ni * C2c + oc) * HWc + hwo;
      out[idx] = xin[idx] + v;
    }
  }
}

// ---------------------------------------------------------------------------
// Host launcher
// ---------------------------------------------------------------------------
extern "C" void kernel_launch(void* const* d_in, const int* in_sizes, int n_in,
                              void* d_out, int out_size, void* d_ws, size_t ws_size,
                              hipStream_t stream)
{
  const float* x     = (const float*)d_in[0];
  const float* w1    = (const float*)d_in[1];
  const float* g1    = (const float*)d_in[2];
  const float* b1    = (const float*)d_in[3];
  const float* m1    = (const float*)d_in[4];
  const float* v1    = (const float*)d_in[5];
  const float* w2    = (const float*)d_in[6];
  const float* g2    = (const float*)d_in[7];
  const float* b2    = (const float*)d_in[8];
  const float* m2    = (const float*)d_in[9];
  const float* v2    = (const float*)d_in[10];
  const float* dw_w  = (const float*)d_in[11];
  const float* dw_b  = (const float*)d_in[12];
  const float* ln_g  = (const float*)d_in[13];
  const float* ln_b  = (const float*)d_in[14];
  const float* off_w = (const float*)d_in[15];
  const float* off_b = (const float*)d_in[16];
  const float* msk_w = (const float*)d_in[17];
  const float* msk_b = (const float*)d_in[18];
  const float* in_w  = (const float*)d_in[19];
  const float* in_b  = (const float*)d_in[20];
  const float* out_w = (const float*)d_in[21];
  const float* out_b = (const float*)d_in[22];
  const float* g3    = (const float*)d_in[23];
  const float* b3    = (const float*)d_in[24];
  const float* m3    = (const float*)d_in[25];
  const float* v3    = (const float*)d_in[26];

  char* ws = (char*)d_ws;
  _Float16* y1mk  = (_Float16*)(ws);                        //  6,553,600 B
  float*    ynhw  = (float*)(ws + 6553600);                 // 26,214,400 B
  float*    xprj  = (float*)(ws + 32768000);                // 26,214,400 B
  float*    zbuf  = (float*)(ws + 58982400);                // 26,214,400 B
  float*    offb  = (float*)(ws + 85196800);                //  1,843,200 B
  float*    mskb  = (float*)(ws + 87040000);                //    921,600 B
  float*    xnhwc = (float*)(ws + 87961600);                // 26,214,400 B
  _Float16* wpk   = (_Float16*)(ws + 114176000);            //    589,824 B

  dim3 blk(256);
  // prep: x -> NHWC (LDS-tiled transpose), w1 -> f16 tap-major
  k_x_to_nhwc<<<dim3(HWc / 32, C1c / 32, NB), blk, 0, stream>>>(x, xnhwc);
  k_pack_w1<<<(CHc * K1c + 255) / 256, blk, 0, stream>>>(w1, wpk);
  // cv1: 1600 M-tiles x 2 N-groups = 3200 waves / 8 per block
  k_cv1<<<400, blk, 0, stream>>>(xnhwc, wpk, g1, b1, m1, v1, y1mk);
  // cv2: 1600 x 4 groups = 6400 waves
  k_cv2<<<800, blk, 0, stream>>>(y1mk, w2, g2, b2, m2, v2, ynhw);
  // dw + LN + GELU + offset/mask: one block per pixel
  k_dw_ln_gelu<<<MPIX, blk, 0, stream>>>(ynhw, dw_w, dw_b, ln_g, ln_b,
                                         off_w, off_b, msk_w, msk_b, offb, mskb);
  // input_proj: 6400 waves
  k_inproj<<<800, blk, 0, stream>>>(ynhw, in_w, in_b, xprj);
  // DCNv3 core: one block per pixel
  k_dcnv3<<<MPIX, blk, 0, stream>>>(xprj, offb, mskb, zbuf);
  // output_proj + BN + SiLU + residual: 6400 waves
  k_outproj<<<800, blk, 0, stream>>>(zbuf, out_w, out_b, g3, b3, m3, v3,
                                     x, (float*)d_out);
}
